// Block_65180423684994
// MI455X (gfx1250) — compile-verified, matrix operand order checked by reading, not run on hardware
//
#include <hip/hip_runtime.h>
#include <hip/hip_bf16.h>
#include <math.h>

// ---------------------------------------------------------------------------
// Types
// ---------------------------------------------------------------------------
typedef __bf16 bf16;
typedef __attribute__((ext_vector_type(16))) __bf16 v16bf;
typedef __attribute__((ext_vector_type(8)))  __bf16 v8bf;   // 16 bytes
typedef __attribute__((ext_vector_type(4)))  __bf16 v4bf;   // 8 bytes
typedef __attribute__((ext_vector_type(8)))  float  v8f;

#define CC   256
#define GG   16
#define CGRP 16          // channels per group
#define PP   9
#define HH   56
#define WW_  56
#define NN   8
#define LTOK (NN*HH*WW_) // 25088
#define HID  1024
#define NOFFM 432        // 288 offsets + 144 mask logits (logical)
#define NOFFM_PAD 448    // padded to multiple of 64 so B loads need no guard

// ---------------------------------------------------------------------------
// bf16 fragment loader (A 16x32 and B(=W row-major NxK) 32x16 share the
// same per-lane pattern on CDNA5: lane<16 -> rows 0..15, K {0..7,16..23};
// lane>=16 -> K {8..15,24..31}; elements 2r,2r+1 live in VGPR r)
// ---------------------------------------------------------------------------
__device__ __forceinline__ v16bf load_tile(const bf16* __restrict__ base,
                                           int ld, int lane) {
    int r  = lane & 15;
    int kh = (lane >> 4) << 3;
    const bf16* p = base + (size_t)r * ld + kh;
    v8bf lo = *(const v8bf*)(p);
    v8bf hi = *(const v8bf*)(p + 16);
    v16bf f;
#pragma unroll
    for (int i = 0; i < 8; ++i) { f[i] = lo[i]; f[i + 8] = hi[i]; }
    return f;
}

// ---------------------------------------------------------------------------
// Generic WMMA GEMM:  out = fuse( A(MxK) * W(NxK)^T + bias ) [+ residual]
// wave tile: 32(M) x 64(N) = 8 accumulators; block = 4 waves = 128x64 tile.
// Preconditions: Kd % 32 == 0; M % 128 == 0; W has >= gridDim.y*64 rows
// (zero-padded past Nd). Store/bias guarded by logical Nd (also output ld).
// fuse: 0 = none, 1 = exact GELU
// ---------------------------------------------------------------------------
__global__ void __launch_bounds__(128)
wmma_gemm_kernel(const bf16* __restrict__ A, const bf16* __restrict__ W,
                 const float* __restrict__ bias, const float* __restrict__ residual,
                 float* __restrict__ outF, bf16* __restrict__ outB,
                 int M, int Kd, int Nd, int fuse) {
    int lane = threadIdx.x & 31;
    int wave = threadIdx.x >> 5;
    int m0 = blockIdx.x * 128 + wave * 32;
    int n0 = blockIdx.y * 64;
    if (m0 >= M) return;

    v8f acc[2][4];
#pragma unroll
    for (int i = 0; i < 2; ++i)
#pragma unroll
        for (int j = 0; j < 4; ++j)
#pragma unroll
            for (int r = 0; r < 8; ++r) acc[i][j][r] = 0.0f;

    const bf16* A0 = A + (size_t)m0 * Kd;
    const bf16* A1 = A + (size_t)(m0 + 16) * Kd;
    const bf16* W0 = W + (size_t)(n0 +  0) * Kd;
    const bf16* W1 = W + (size_t)(n0 + 16) * Kd;
    const bf16* W2 = W + (size_t)(n0 + 32) * Kd;
    const bf16* W3 = W + (size_t)(n0 + 48) * Kd;

    // ---- software-pipelined K loop (2 stages) ----
    v16bf a0 = load_tile(A0, Kd, lane);
    v16bf a1 = load_tile(A1, Kd, lane);
    v16bf b0 = load_tile(W0, Kd, lane);
    v16bf b1 = load_tile(W1, Kd, lane);
    v16bf b2 = load_tile(W2, Kd, lane);
    v16bf b3 = load_tile(W3, Kd, lane);

    for (int k0 = 32; k0 < Kd; k0 += 32) {
        __builtin_prefetch(A0 + k0 + 32, 0, 3);   // near-cache prefetch
        v16bf na0 = load_tile(A0 + k0, Kd, lane);
        v16bf na1 = load_tile(A1 + k0, Kd, lane);
        v16bf nb0 = load_tile(W0 + k0, Kd, lane);
        v16bf nb1 = load_tile(W1 + k0, Kd, lane);
        v16bf nb2 = load_tile(W2 + k0, Kd, lane);
        v16bf nb3 = load_tile(W3 + k0, Kd, lane);

        acc[0][0] = __builtin_amdgcn_wmma_f32_16x16x32_bf16(false, a0, false, b0, (short)0, acc[0][0], false, false);
        acc[0][1] = __builtin_amdgcn_wmma_f32_16x16x32_bf16(false, a0, false, b1, (short)0, acc[0][1], false, false);
        acc[0][2] = __builtin_amdgcn_wmma_f32_16x16x32_bf16(false, a0, false, b2, (short)0, acc[0][2], false, false);
        acc[0][3] = __builtin_amdgcn_wmma_f32_16x16x32_bf16(false, a0, false, b3, (short)0, acc[0][3], false, false);
        acc[1][0] = __builtin_amdgcn_wmma_f32_16x16x32_bf16(false, a1, false, b0, (short)0, acc[1][0], false, false);
        acc[1][1] = __builtin_amdgcn_wmma_f32_16x16x32_bf16(false, a1, false, b1, (short)0, acc[1][1], false, false);
        acc[1][2] = __builtin_amdgcn_wmma_f32_16x16x32_bf16(false, a1, false, b2, (short)0, acc[1][2], false, false);
        acc[1][3] = __builtin_amdgcn_wmma_f32_16x16x32_bf16(false, a1, false, b3, (short)0, acc[1][3], false, false);

        a0 = na0; a1 = na1; b0 = nb0; b1 = nb1; b2 = nb2; b3 = nb3;
    }
    // tail
    acc[0][0] = __builtin_amdgcn_wmma_f32_16x16x32_bf16(false, a0, false, b0, (short)0, acc[0][0], false, false);
    acc[0][1] = __builtin_amdgcn_wmma_f32_16x16x32_bf16(false, a0, false, b1, (short)0, acc[0][1], false, false);
    acc[0][2] = __builtin_amdgcn_wmma_f32_16x16x32_bf16(false, a0, false, b2, (short)0, acc[0][2], false, false);
    acc[0][3] = __builtin_amdgcn_wmma_f32_16x16x32_bf16(false, a0, false, b3, (short)0, acc[0][3], false, false);
    acc[1][0] = __builtin_amdgcn_wmma_f32_16x16x32_bf16(false, a1, false, b0, (short)0, acc[1][0], false, false);
    acc[1][1] = __builtin_amdgcn_wmma_f32_16x16x32_bf16(false, a1, false, b1, (short)0, acc[1][1], false, false);
    acc[1][2] = __builtin_amdgcn_wmma_f32_16x16x32_bf16(false, a1, false, b2, (short)0, acc[1][2], false, false);
    acc[1][3] = __builtin_amdgcn_wmma_f32_16x16x32_bf16(false, a1, false, b3, (short)0, acc[1][3], false, false);

    // C/D layout: VGPR r, lanes 0..15 -> (M=r, N=lane); lanes 16..31 -> (M=r+8)
    int nlane  = lane & 15;
    int rowoff = (lane >> 4) * 8;
#pragma unroll
    for (int j = 0; j < 4; ++j) {
        int colg = n0 + 16 * j + nlane;
        if (colg >= Nd) continue;
        float bv = bias ? bias[colg] : 0.0f;
#pragma unroll
        for (int i = 0; i < 2; ++i) {
#pragma unroll
            for (int r = 0; r < 8; ++r) {
                int rowg = m0 + 16 * i + rowoff + r;
                float v = acc[i][j][r] + bv;
                if (fuse == 1) v = 0.5f * v * (1.0f + erff(v * 0.70710678118654752f));
                size_t o = (size_t)rowg * Nd + colg;
                if (residual) v += residual[o];
                if (outF) outF[o] = v;
                if (outB) outB[o] = (bf16)v;
            }
        }
    }
}

// ---------------------------------------------------------------------------
// LayerNorm over C=256, fp32 in -> bf16 out. One wave per token.
// ---------------------------------------------------------------------------
__global__ void __launch_bounds__(256)
ln_kernel(const float* __restrict__ x, const float* __restrict__ gg,
          const float* __restrict__ bb, bf16* __restrict__ out, int L) {
    int lane = threadIdx.x & 31;
    int wave = threadIdx.x >> 5;
    int t = blockIdx.x * 8 + wave;
    if (t >= L) return;
    const float4* xp4 = (const float4*)(x + (size_t)t * CC);
    float4 v0 = xp4[lane];
    float4 v1 = xp4[lane + 32];
    float s  = v0.x + v0.y + v0.z + v0.w + v1.x + v1.y + v1.z + v1.w;
    float ss = v0.x*v0.x + v0.y*v0.y + v0.z*v0.z + v0.w*v0.w
             + v1.x*v1.x + v1.y*v1.y + v1.z*v1.z + v1.w*v1.w;
#pragma unroll
    for (int m = 16; m >= 1; m >>= 1) {
        s  += __shfl_xor(s,  m, 32);
        ss += __shfl_xor(ss, m, 32);
    }
    float mu  = s * (1.0f / CC);
    float inv = rsqrtf(ss * (1.0f / CC) - mu * mu + 1e-6f);
    int c0 = lane * 4, c1 = 128 + lane * 4;
    bf16* o = out + (size_t)t * CC;
    v4bf o0, o1;
    o0[0] = (bf16)((v0.x - mu) * inv * gg[c0 + 0] + bb[c0 + 0]);
    o0[1] = (bf16)((v0.y - mu) * inv * gg[c0 + 1] + bb[c0 + 1]);
    o0[2] = (bf16)((v0.z - mu) * inv * gg[c0 + 2] + bb[c0 + 2]);
    o0[3] = (bf16)((v0.w - mu) * inv * gg[c0 + 3] + bb[c0 + 3]);
    o1[0] = (bf16)((v1.x - mu) * inv * gg[c1 + 0] + bb[c1 + 0]);
    o1[1] = (bf16)((v1.y - mu) * inv * gg[c1 + 1] + bb[c1 + 1]);
    o1[2] = (bf16)((v1.z - mu) * inv * gg[c1 + 2] + bb[c1 + 2]);
    o1[3] = (bf16)((v1.w - mu) * inv * gg[c1 + 3] + bb[c1 + 3]);
    *(v4bf*)(o + c0) = o0;
    *(v4bf*)(o + c1) = o1;
}

// ---------------------------------------------------------------------------
// Depthwise 3x3 conv (pad 1), bf16 in/out, fp32 accumulate. Block = 1 token.
// ---------------------------------------------------------------------------
__global__ void __launch_bounds__(256)
dwconv_kernel(const bf16* __restrict__ lnx, const float* __restrict__ dw_w,
              const float* __restrict__ dw_b, bf16* __restrict__ x1) {
    int t = blockIdx.x;
    int c = threadIdx.x;
    int n  = t / (HH * WW_);
    int hw = t - n * HH * WW_;
    int h  = hw / WW_;
    int w  = hw - h * WW_;
    float acc = dw_b[c];
#pragma unroll
    for (int ky = 0; ky < 3; ++ky) {
#pragma unroll
        for (int kx = 0; kx < 3; ++kx) {
            int yy = h + ky - 1, xx = w + kx - 1;
            if (yy >= 0 && yy < HH && xx >= 0 && xx < WW_) {
                size_t idx = ((size_t)((n * HH + yy) * WW_ + xx)) * CC + c;
                acc += (float)lnx[idx] * dw_w[(ky * 3 + kx) * CC + c];
            }
        }
    }
    x1[(size_t)t * CC + c] = (bf16)acc;
}

// ---------------------------------------------------------------------------
// DCNv3 sampling core: per token, stage 432 offset/mask values in LDS,
// softmax per group, 9-point bilinear gather from xp (fp32), write bf16.
// Thread (g = tid>>4, c = tid&15).
// ---------------------------------------------------------------------------
__global__ void __launch_bounds__(256)
dcn_sample_kernel(const float* __restrict__ xp, const float* __restrict__ om,
                  bf16* __restrict__ outc) {
    __shared__ float sm[NOFFM];
    int t   = blockIdx.x;
    int tid = threadIdx.x;
    const float* omt = om + (size_t)t * NOFFM;
    if (tid < NOFFM)       sm[tid]       = omt[tid];
    if (tid + 256 < NOFFM) sm[tid + 256] = omt[tid + 256];
    __syncthreads();

    int g = tid >> 4, c = tid & 15;
    int n  = t / (HH * WW_);
    int hw = t - n * HH * WW_;
    int h  = hw / WW_;
    int w  = hw - h * WW_;

    // softmax over the 9 mask logits of this group
    float e[PP];
    float mx = -1e30f;
#pragma unroll
    for (int p = 0; p < PP; ++p) {
        float l = sm[288 + g * PP + p];
        mx = fmaxf(mx, l);
        e[p] = l;
    }
    float se = 0.0f;
#pragma unroll
    for (int p = 0; p < PP; ++p) { e[p] = __expf(e[p] - mx); se += e[p]; }
    float invse = 1.0f / se;

    const float invD = 1.0f / 58.0f;              // Hin = Win = 58
    float refx = (w + 1.5f) * invD;
    float refy = (h + 1.5f) * invD;

    const float* xc = xp + (size_t)(n * HH * WW_) * CC + g * CGRP + c;
    float accv = 0.0f;
#pragma unroll
    for (int p = 0; p < PP; ++p) {
        int pi = p / 3, pj = p - 3 * pi;           // x-major grid per torch ref
        float lx = refx + (pi - 1) * invD + sm[g * 18 + p * 2 + 0] * invD;
        float ly = refy + (pj - 1) * invD + sm[g * 18 + p * 2 + 1] * invD;
        float ox = lx * 58.0f - 1.5f;              // padded->orig coords folded
        float oy = ly * 58.0f - 1.5f;
        float x0f = floorf(ox), y0f = floorf(oy);
        float wx = ox - x0f, wy = oy - y0f;
        int ix0 = (int)x0f, iy0 = (int)y0f;
        float s = 0.0f;
#pragma unroll
        for (int dy = 0; dy < 2; ++dy) {
#pragma unroll
            for (int dx = 0; dx < 2; ++dx) {
                int ix = ix0 + dx, iy = iy0 + dy;
                if (ix >= 0 && ix < WW_ && iy >= 0 && iy < HH) {
                    float wgt = (dx ? wx : 1.0f - wx) * (dy ? wy : 1.0f - wy);
                    s += wgt * xc[((size_t)iy * WW_ + ix) * CC];
                }
            }
        }
        accv += e[p] * s;
    }
    outc[(size_t)t * CC + tid] = (bf16)(accv * invse);
}

// ---------------------------------------------------------------------------
// Weight conversion helpers
// ---------------------------------------------------------------------------
__global__ void f32_to_bf16_kernel(const float* __restrict__ in,
                                   bf16* __restrict__ out, int n) {
    int i = blockIdx.x * 256 + threadIdx.x;
    if (i < n) out[i] = (bf16)in[i];
}

// builds zero-padded 448x256 fused [offset;mask] weight + 432 fused bias
__global__ void build_comb_kernel(const float* __restrict__ off_w,
                                  const float* __restrict__ mask_w,
                                  const float* __restrict__ off_b,
                                  const float* __restrict__ mask_b,
                                  bf16* __restrict__ wcomb,
                                  float* __restrict__ bcomb) {
    int i = blockIdx.x * 256 + threadIdx.x;
    const int total = NOFFM_PAD * CC;
    if (i < total) {
        int r = i >> 8, cc = i & 255;
        float v = 0.0f;
        if (r < 288)        v = off_w[r * CC + cc];
        else if (r < NOFFM) v = mask_w[(r - 288) * CC + cc];
        wcomb[i] = (bf16)v;
    }
    if (i < NOFFM) bcomb[i] = (i < 288) ? off_b[i] : mask_b[i - 288];
}

// ---------------------------------------------------------------------------
// Host: orchestrate the whole block on `stream`
// ---------------------------------------------------------------------------
extern "C" void kernel_launch(void* const* d_in, const int* in_sizes, int n_in,
                              void* d_out, int out_size, void* d_ws, size_t ws_size,
                              hipStream_t stream) {
    const float* x      = (const float*)d_in[0];
    const float* ln1_g  = (const float*)d_in[1];
    const float* ln1_b  = (const float*)d_in[2];
    const float* dw_w   = (const float*)d_in[3];
    const float* dw_b   = (const float*)d_in[4];
    const float* off_w  = (const float*)d_in[5];
    const float* off_b  = (const float*)d_in[6];
    const float* mask_w = (const float*)d_in[7];
    const float* mask_b = (const float*)d_in[8];
    const float* inp_w  = (const float*)d_in[9];
    const float* inp_b  = (const float*)d_in[10];
    const float* out_w  = (const float*)d_in[11];
    const float* out_b  = (const float*)d_in[12];
    const float* ln2_g  = (const float*)d_in[13];
    const float* ln2_b  = (const float*)d_in[14];
    const float* fc1_w  = (const float*)d_in[15];
    const float* fc1_b  = (const float*)d_in[16];
    const float* fc2_w  = (const float*)d_in[17];
    const float* fc2_b  = (const float*)d_in[18];
    float* out = (float*)d_out;

    const size_t L = LTOK;

    // workspace carve (overlaid buffers, ~148 MB total)
    char* p = (char*)d_ws;
    auto carve = [&](size_t bytes) -> void* {
        void* r = (void*)p;
        p += (bytes + 255) & ~(size_t)255;
        return r;
    };
    bf16*  buf1 = (bf16*)carve(L * CC * sizeof(bf16));     // lnx, later hb
    bf16*  buf2 = (bf16*)carve(L * HID * sizeof(bf16));    // x1 (front), later gb
    float* buf3 = (float*)carve(L * CC * sizeof(float));   // xp, later y
    float* om   = (float*)carve(L * NOFFM * sizeof(float));
    bf16*  dcn  = (bf16*)carve(L * CC * sizeof(bf16));
    bf16*  w_inp = (bf16*)carve(CC * CC * sizeof(bf16));
    bf16*  w_out = (bf16*)carve(CC * CC * sizeof(bf16));
    bf16*  w_fc1 = (bf16*)carve(HID * CC * sizeof(bf16));
    bf16*  w_fc2 = (bf16*)carve(CC * HID * sizeof(bf16));
    bf16*  w_cmb = (bf16*)carve(NOFFM_PAD * CC * sizeof(bf16));
    float* b_cmb = (float*)carve(NOFFM * sizeof(float));

    bf16* lnx = buf1;
    bf16* hb  = buf1;
    bf16* x1  = buf2;
    bf16* gb  = buf2;
    float* xp = buf3;
    float* y  = buf3;

    // weight conversion (tiny)
    f32_to_bf16_kernel<<<(CC * CC + 255) / 256, 256, 0, stream>>>(inp_w, w_inp, CC * CC);
    f32_to_bf16_kernel<<<(CC * CC + 255) / 256, 256, 0, stream>>>(out_w, w_out, CC * CC);
    f32_to_bf16_kernel<<<(HID * CC + 255) / 256, 256, 0, stream>>>(fc1_w, w_fc1, HID * CC);
    f32_to_bf16_kernel<<<(CC * HID + 255) / 256, 256, 0, stream>>>(fc2_w, w_fc2, CC * HID);
    build_comb_kernel<<<(NOFFM_PAD * CC + 255) / 256, 256, 0, stream>>>(off_w, mask_w, off_b, mask_b, w_cmb, b_cmb);

    const int MB = (int)(L / 128);                 // 196 macro M-tiles
    dim3 blk(128);

    // LN1 -> bf16
    ln_kernel<<<(int)(L / 8), 256, 0, stream>>>(x, ln1_g, ln1_b, lnx, (int)L);
    // depthwise 3x3
    dwconv_kernel<<<(int)L, 256, 0, stream>>>(lnx, dw_w, dw_b, x1);
    // input_proj: xp = lnx @ inp_w^T + b  (fp32 out for sampling)
    wmma_gemm_kernel<<<dim3(MB, 4), blk, 0, stream>>>(lnx, w_inp, inp_b, nullptr,
                                                      xp, nullptr, (int)L, CC, CC, 0);
    // fused offsets+mask: om = x1 @ [off;mask]^T + b (weights zero-padded to 448)
    wmma_gemm_kernel<<<dim3(MB, 7), blk, 0, stream>>>(x1, w_cmb, b_cmb, nullptr,
                                                      om, nullptr, (int)L, CC, NOFFM, 0);
    // DCNv3 sampling core
    dcn_sample_kernel<<<(int)L, 256, 0, stream>>>(xp, om, dcn);
    // output_proj + residual(x): y = x + dcn @ out_w^T + b
    wmma_gemm_kernel<<<dim3(MB, 4), blk, 0, stream>>>(dcn, w_out, out_b, x,
                                                      y, nullptr, (int)L, CC, CC, 0);
    // LN2 -> bf16
    ln_kernel<<<(int)(L / 8), 256, 0, stream>>>(y, ln2_g, ln2_b, hb, (int)L);
    // fc1 + exact GELU -> bf16
    wmma_gemm_kernel<<<dim3(MB, 16), blk, 0, stream>>>(hb, w_fc1, fc1_b, nullptr,
                                                       nullptr, gb, (int)L, CC, HID, 1);
    // fc2 + residual(y) -> d_out (fp32)
    wmma_gemm_kernel<<<dim3(MB, 4), blk, 0, stream>>>(gb, w_fc2, fc2_b, y,
                                                      out, nullptr, (int)L, HID, CC, 0);
}
